// PLELayer_36799279792799
// MI455X (gfx1250) — matrix operand order repaired and checked by reading, not compiled, not run on hardware
//
#include <hip/hip_runtime.h>
#include <hip/hip_bf16.h>

// Problem constants (from reference): T=2, E=8, D=1024, O=1024, B=4096
#define T_TASKS 2
#define NE      8
#define DIMK    1024
#define ODIM    1024
#define BATCH   4096
#define NCOLS   (NE * ODIM)          // 8192 expert-output columns (o*E + e)

// Tiling for the fused MoE GEMM
#define BM 128                        // batch rows per workgroup
#define BN 128                        // expert columns per workgroup (= 16 o * 8 e)
#define BK 32                         // K step (one bf16 WMMA K)
#define A_STRIDE 20                   // dwords per A-tile row (16 data + 4 pad)
#define B_STRIDE 65                   // dwords per B-tile row (64 data + 1 pad)

typedef __bf16 v16bf __attribute__((ext_vector_type(16)));
typedef __bf16 v2bf  __attribute__((ext_vector_type(2)));
typedef float  v8f   __attribute__((ext_vector_type(8)));

union WFrag {
    v16bf    v;
    unsigned u[8];
};

// RNE f32 -> bf16 pair packed into one dword (lo = first element).
__device__ __forceinline__ unsigned bf16pair(float lo, float hi) {
    v2bf p;
    p.x = (__bf16)lo;
    p.y = (__bf16)hi;
    return __builtin_bit_cast(unsigned, p);
}

// ---------------------------------------------------------------------------
// Kernel 1: gate logits (exact fp32).  ws layout:
//   gsL: [B][24]   share-gate logits (experts ordered: share8, task0_8, task1_8)
//   gtL: [T][B][16] task-gate logits  (share-half 0..7, task-half 8..15)
// ---------------------------------------------------------------------------
__global__ __launch_bounds__(256) void ple_gate_logits_kernel(
    const float* __restrict__ xs,   // [B, D]
    const float* __restrict__ xt,   // [T, B, D]
    const float* __restrict__ tgk,  // [T, D, 16]
    const float* __restrict__ sgk,  // [D, 24]
    float* __restrict__ gsL, float* __restrict__ gtL)
{
    int gid = blockIdx.x * 256 + threadIdx.x;
    if (gid >= BATCH * 56) return;
    int b = gid / 56, c = gid % 56;
    float acc = 0.0f;
    if (c < 24) {
        const float* x = xs + (size_t)b * DIMK;
        #pragma unroll 8
        for (int k = 0; k < DIMK; ++k) acc += x[k] * sgk[(size_t)k * 24 + c];
        gsL[b * 24 + c] = acc;
    } else {
        int c2 = c - 24, t = c2 >> 4, g = c2 & 15;
        const float* x = xt + ((size_t)t * BATCH + b) * DIMK;
        const float* w = tgk + (size_t)t * DIMK * 16;
        #pragma unroll 8
        for (int k = 0; k < DIMK; ++k) acc += x[k] * w[k * 16 + g];
        gtL[(t * BATCH + b) * 16 + g] = acc;
    }
}

// ---------------------------------------------------------------------------
// Kernel 2: in-place softmax of the gate logits (one thread per batch row).
// ---------------------------------------------------------------------------
__global__ __launch_bounds__(256) void ple_softmax_kernel(
    float* __restrict__ gs, float* __restrict__ gt)
{
    int b = blockIdx.x * 256 + threadIdx.x;
    if (b >= BATCH) return;
    {
        float* p = gs + (size_t)b * 24;
        float m = p[0];
        #pragma unroll
        for (int i = 1; i < 24; ++i) m = fmaxf(m, p[i]);
        float s = 0.0f;
        #pragma unroll
        for (int i = 0; i < 24; ++i) { float e = __expf(p[i] - m); p[i] = e; s += e; }
        float inv = 1.0f / s;
        #pragma unroll
        for (int i = 0; i < 24; ++i) p[i] *= inv;
    }
    #pragma unroll
    for (int t = 0; t < T_TASKS; ++t) {
        float* p = gt + ((size_t)t * BATCH + b) * 16;
        float m = p[0];
        #pragma unroll
        for (int i = 1; i < 16; ++i) m = fmaxf(m, p[i]);
        float s = 0.0f;
        #pragma unroll
        for (int i = 0; i < 16; ++i) { float e = __expf(p[i] - m); p[i] = e; s += e; }
        float inv = 1.0f / s;
        #pragma unroll
        for (int i = 0; i < 16; ++i) p[i] *= inv;
    }
}

// ---------------------------------------------------------------------------
// Kernel 3: fused expert GEMMs (bf16 WMMA, f32 accumulate) + gated combine.
// Grid: (NCOLS/BN, BATCH/BM); 8 waves (256 threads); wave tile 64x32.
// Sources: src 0 = shared experts, src 1/2 = task experts.
// ---------------------------------------------------------------------------
__global__ __launch_bounds__(256, 4) void ple_moe_kernel(
    const float* __restrict__ xs,     // [B, D]
    const float* __restrict__ xt,     // [T, B, D]
    const float* __restrict__ wS,     // [D, 8192]
    const float* __restrict__ wT,     // [T, D, 8192]
    const float* __restrict__ gatesS, // [B, 24]   softmaxed
    const float* __restrict__ gatesT, // [T, B, 16] softmaxed
    float* __restrict__ outS,         // [B, O]
    float* __restrict__ outT)         // [T, B, O]
{
    __shared__ __align__(16) unsigned Ald[BM * A_STRIDE]; // bf16 A tile (pairs)
    __shared__ __align__(16) unsigned Bld[BK * B_STRIDE]; // bf16 B tile (pairs)
    __shared__ float oS[BM * 16];
    __shared__ float oT[T_TASKS][BM * 16];

    const int tid  = threadIdx.x;
    const int lane = tid & 31;
    const int wave = tid >> 5;                 // 0..7
    const int wm0  = (wave & 1) * 64;          // wave row base in tile (2 m-waves)
    const int wn0  = (wave >> 1) * 32;         // wave col base in tile (4 n-waves)
    const int row0 = blockIdx.y * BM;          // global batch row base
    const int col0 = blockIdx.x * BN;          // global expert-column base
    const int o0   = col0 / NE;                // global o base (16 o per tile)

    const int hi16 = lane >> 4;                // 0/1: which half of the wave
    const int m16  = lane & 15;
    const int e    = lane & 7;                 // expert id within group
    const int og   = (lane >> 3) & 1;          // o offset within a 16-col subtile

    for (int i = tid; i < BM * 16; i += 256) {
        oS[i] = 0.0f; oT[0][i] = 0.0f; oT[1][i] = 0.0f;
    }
    __syncthreads();

    #pragma unroll 1
    for (int src = 0; src < 3; ++src) {
        const float* Ap = (src == 0) ? xs : xt + (size_t)(src - 1) * BATCH * DIMK;
        const float* Bp = (src == 0) ? wS : wT + (size_t)(src - 1) * DIMK * NCOLS;

        v8f acc[4][2] = {};
        float4 pa[4], pb[4];

        auto loadA = [&](int kt) {
            #pragma unroll
            for (int it = 0; it < 4; ++it) {
                int f4 = it * 256 + tid;           // 0..1023
                int r = f4 >> 3, c4 = f4 & 7;      // 128 rows x 8 float4
                pa[it] = *(const float4*)(Ap + (size_t)(row0 + r) * DIMK + kt * BK + c4 * 4);
            }
        };
        auto loadB = [&](int kt) {
            #pragma unroll
            for (int it = 0; it < 4; ++it) {
                int f4 = it * 256 + tid;
                int r = f4 >> 5, c4 = f4 & 31;     // 32 rows x 32 float4
                pb[it] = *(const float4*)(Bp + (size_t)(kt * BK + r) * NCOLS + col0 + c4 * 4);
            }
        };
        auto storeAB = [&]() {
            #pragma unroll
            for (int it = 0; it < 4; ++it) {
                int f4 = it * 256 + tid;
                int r = f4 >> 3, c4 = f4 & 7;
                Ald[r * A_STRIDE + c4 * 2    ] = bf16pair(pa[it].x, pa[it].y);
                Ald[r * A_STRIDE + c4 * 2 + 1] = bf16pair(pa[it].z, pa[it].w);
            }
            #pragma unroll
            for (int it = 0; it < 4; ++it) {
                int f4 = it * 256 + tid;
                int r = f4 >> 5, c4 = f4 & 31;
                Bld[r * B_STRIDE + c4 * 2    ] = bf16pair(pb[it].x, pb[it].y);
                Bld[r * B_STRIDE + c4 * 2 + 1] = bf16pair(pb[it].z, pb[it].w);
            }
        };

        loadA(0); loadB(0);
        #pragma unroll 1
        for (int kt = 0; kt < DIMK / BK; ++kt) {
            __syncthreads();
            storeAB();
            __syncthreads();
            if (kt + 1 < DIMK / BK) { loadA(kt + 1); loadB(kt + 1); }

            // Build fragments from LDS.
            // A (16x32 bf16): lane m = lane&15, K pairs: v<4 -> K=2v(,+1), v>=4 -> K=16+2(v-4);
            // +8 for lanes 16..31.  Dword layout is two aligned uint4 groups.
            WFrag a[4], b[2];
            #pragma unroll
            for (int mi = 0; mi < 4; ++mi) {
                int m = wm0 + mi * 16 + m16;
                const uint4* p = (const uint4*)&Ald[m * A_STRIDE + hi16 * 4];
                uint4 q0 = p[0];
                uint4 q1 = p[2];                   // +8 dwords
                a[mi].u[0] = q0.x; a[mi].u[1] = q0.y; a[mi].u[2] = q0.z; a[mi].u[3] = q0.w;
                a[mi].u[4] = q1.x; a[mi].u[5] = q1.y; a[mi].u[6] = q1.z; a[mi].u[7] = q1.w;
            }
            // B (32x16 bf16): lane n = lane&15, K pairs same as A: v<4 -> K=2v(,+1), v>=4 -> K=16+2(v-4); +8 for lanes 16..31.
            #pragma unroll
            for (int ni = 0; ni < 2; ++ni) {
                int n = wn0 + ni * 16 + m16;
                const unsigned* p = &Bld[n >> 1];
                int sh = (n & 1) * 16;
                #pragma unroll
                for (int v = 0; v < 8; ++v) {
                    int k0 = (v < 4 ? 2 * v : 16 + 2 * (v - 4)) + hi16 * 8;
                    unsigned lo = (p[k0 * B_STRIDE] >> sh) & 0xffffu;
                    unsigned hi = (p[(k0 + 1) * B_STRIDE] >> sh) & 0xffffu;
                    b[ni].u[v] = lo | (hi << 16);
                }
            }
            #pragma unroll
            for (int mi = 0; mi < 4; ++mi)
                #pragma unroll
                for (int ni = 0; ni < 2; ++ni)
                    acc[mi][ni] = __builtin_amdgcn_wmma_f32_16x16x32_bf16(
                        false, a[mi].v, false, b[ni].v, (short)0, acc[mi][ni],
                        false, false);
        }

        // Gated fold: C element (VGPR r) is row wm0+mi*16+r+hi16*8, col wn0+ni*16+(lane&15).
        // Reduce over the 8-expert lane groups with LDS float atomics.
        #pragma unroll 1
        for (int mi = 0; mi < 4; ++mi) {
            #pragma unroll
            for (int r = 0; r < 8; ++r) {
                int rl = wm0 + mi * 16 + r + hi16 * 8;  // tile-local row
                int rg = row0 + rl;                     // global batch row
                float gS = gatesS[rg * 24 + src * 8 + e];
                float gA = 0.0f, gB = 0.0f;
                if (src == 0) {
                    gA = gatesT[(size_t)(0 * BATCH + rg) * 16 + e];
                    gB = gatesT[(size_t)(1 * BATCH + rg) * 16 + e];
                } else {
                    gA = gatesT[(size_t)((src - 1) * BATCH + rg) * 16 + 8 + e];
                }
                #pragma unroll
                for (int ni = 0; ni < 2; ++ni) {
                    int ol = ((wn0 + ni * 16) >> 3) + og;   // tile-local o (0..15)
                    float v = acc[mi][ni][r];
                    atomicAdd(&oS[rl * 16 + ol], v * gS);
                    if (src == 0) {
                        atomicAdd(&oT[0][rl * 16 + ol], v * gA);
                        atomicAdd(&oT[1][rl * 16 + ol], v * gB);
                    } else {
                        atomicAdd(&oT[src - 1][rl * 16 + ol], v * gA);
                    }
                }
            }
        }
    }

    __syncthreads();
    for (int i = tid; i < BM * 16; i += 256) {
        int rl = i >> 4, ol = i & 15;
        size_t rg = row0 + rl;
        int o = o0 + ol;
        outS[rg * ODIM + o] = oS[i];
        outT[(0 * (size_t)BATCH + rg) * ODIM + o] = oT[0][i];
        outT[(1 * (size_t)BATCH + rg) * ODIM + o] = oT[1][i];
    }
}

// ---------------------------------------------------------------------------
extern "C" void kernel_launch(void* const* d_in, const int* in_sizes, int n_in,
                              void* d_out, int out_size, void* d_ws, size_t ws_size,
                              hipStream_t stream) {
    (void)in_sizes; (void)n_in; (void)out_size; (void)ws_size;
    const float* xs  = (const float*)d_in[0]; // gate_output_shared_final [B,D]
    const float* xt  = (const float*)d_in[1]; // gate_output_task_final   [T,B,D]
    const float* wS  = (const float*)d_in[2]; // share_expert_kernel      [D,E*O]
    const float* wT  = (const float*)d_in[3]; // task_expert_kernel       [T,D,E*O]
    const float* tgk = (const float*)d_in[4]; // task_gate_kernels        [T,D,2E]
    const float* sgk = (const float*)d_in[5]; // share_gate_kernel        [D,(T+1)E]

    float* gsL = (float*)d_ws;                // [B,24]
    float* gtL = gsL + (size_t)BATCH * 24;    // [T,B,16]

    float* outS = (float*)d_out;                       // [B,O]
    float* outT = outS + (size_t)BATCH * ODIM;         // [T,B,O]

    ple_gate_logits_kernel<<<(BATCH * 56 + 255) / 256, 256, 0, stream>>>(
        xs, xt, tgk, sgk, gsL, gtL);
    ple_softmax_kernel<<<(BATCH + 255) / 256, 256, 0, stream>>>(gsL, gtL);

    dim3 grid(NCOLS / BN, BATCH / BM);       // 64 x 32
    ple_moe_kernel<<<grid, 256, 0, stream>>>(
        xs, xt, wS, wT, gsL, gtL, outS, outT);
}
